// MISA_81990925680993
// MI455X (gfx1250) — matrix operand order repaired and checked by reading, not compiled, not run on hardware
//
#include <hip/hip_runtime.h>
#include <math.h>

// ---------------- types -------------------------------------------------
typedef __attribute__((ext_vector_type(16))) _Float16 v16h;
typedef __attribute__((ext_vector_type(8)))  _Float16 v8h;
typedef __attribute__((ext_vector_type(4)))  _Float16 h4;
typedef __attribute__((ext_vector_type(8)))  float    v8f;

#define HD    1024
#define EF    4
#define NH    8
#define DH    128            // head dim = HD/NH
#define BATCH 4096
#define BE    (BATCH*EF)     // 16384
#define EPSLN 1e-5f

enum { MODE_BIAS = 0, MODE_VEC = 1, MODE_RELU = 2, MODE_SIG = 3, MODE_MEAN4 = 4 };

// Combine two 16-byte-aligned 8-half chunks into a WMMA v16h fragment.
__device__ __forceinline__ v16h ld_frag2(const _Float16* p0, const _Float16* p1) {
  v8h a = *(const v8h*)p0;
  v8h b = *(const v8h*)p1;
  v16h r;
#pragma unroll
  for (int i = 0; i < 8; ++i) { r[i] = a[i]; r[i + 8] = b[i]; }
  return r;
}

// ---------------- WMMA GEMM:  Y[M,N] = A[M,K] @ W[N,K]^T (+epilogue) ----
// Block tile 128x128, 256 threads = 8 waves (4 row-waves x 2 col-waves),
// wave tile 32x64 = 2x4 WMMA 16x16 tiles, K stepped by 32 (one f16 WMMA).
// A,W are f32 in HBM; converted to f16 while staging through LDS.
// Software-pipelined: global loads for tile k+1 are issued AFTER the
// barrier (so the __syncthreads fence only waits on DScnt, not LOADcnt)
// and consumed by the convert phase one full WMMA phase later.
// Loads are addressed directly off the kernel-arg pointers (A + offset)
// so clang's addrspace inference emits global_load_b128, not flat_load.
// MODE_MEAN4: output rows averaged in groups of 4 -> Y is [M/4, N].
__global__ __launch_bounds__(256)
void misa_gemm_wmma(const float* __restrict__ A, const float* __restrict__ W,
                    const float* __restrict__ bias, const float* __restrict__ extra,
                    float* __restrict__ Y, int M, int N, int K, int mode)
{
  constexpr int KS  = 40;          // padded LDS row stride in halves (80B)
  constexpr int TSZ = 128 * KS;    // one tile buffer in halves
  __shared__ __align__(16) _Float16 As[2 * TSZ];
  __shared__ __align__(16) _Float16 Bs[2 * TSZ];

  const int tid  = threadIdx.x;
  const int m0   = blockIdx.y * 128;
  const int n0   = blockIdx.x * 128;
  const int lane = tid & 31;
  const int wave = tid >> 5;
  const int wr   = wave >> 1;       // 0..3 row-wave
  const int wc   = wave & 1;        // 0..1 col-wave
  const int lo   = lane & 15;
  const int hi   = lane >> 4;

  // ---- per-thread staging slots: 4 float4 chunks of A and of W ----------
  size_t aoff[4], woff[4];
  int soff[4];
#pragma unroll
  for (int i = 0; i < 4; ++i) {
    const int idx = tid + i * 256;        // 0..1023
    const int row = idx >> 3;             // 0..127
    const int c4  = (idx & 7) << 2;       // 0,4,...,28
    aoff[i] = (size_t)(m0 + row) * K + c4;
    woff[i] = (size_t)(n0 + row) * K + c4;
    soff[i] = row * KS + c4;
  }

  // ---- prologue: loads for tile 0 ---------------------------------------
  float4 ra[4], rb[4];
#pragma unroll
  for (int i = 0; i < 4; ++i) {
    ra[i] = *(const float4*)(A + aoff[i]);
    rb[i] = *(const float4*)(W + woff[i]);
  }

  v8f acc[2][4] = {};
  int buf = 0;
  for (int kk = 0; kk < K; kk += 32) {
    const int bo = buf * TSZ;
    // ---- convert staged registers -> f16 LDS (waits for in-flight loads)
#pragma unroll
    for (int i = 0; i < 4; ++i) {
      h4 ah = { (_Float16)ra[i].x, (_Float16)ra[i].y,
                (_Float16)ra[i].z, (_Float16)ra[i].w };
      *(h4*)(As + bo + soff[i]) = ah;
      h4 wh = { (_Float16)rb[i].x, (_Float16)rb[i].y,
                (_Float16)rb[i].z, (_Float16)rb[i].w };
      *(h4*)(Bs + bo + soff[i]) = wh;
    }
    __syncthreads();

    // ---- issue global loads for next K-tile AFTER the barrier: the fence
    //      then only waits on DScnt, and these loads overlap the WMMAs ----
    if (kk + 32 < K) {
      __builtin_prefetch(A + aoff[0] + kk + 64, 0, 1);   // one tile further
      __builtin_prefetch(W + woff[0] + kk + 64, 0, 1);
#pragma unroll
      for (int i = 0; i < 4; ++i) {
        ra[i] = *(const float4*)(A + aoff[i] + kk + 32);
        rb[i] = *(const float4*)(W + woff[i] + kk + 32);
      }
    }

    // ---- fragments per ISA layout (A 16x32: hi picks K-chunks {0-7,16-23}
    //      vs {8-15,24-31}; B 32x16: same, hi picks K {0-7,16-23} vs {8-15,24-31})
    v16h af[2], bf[4];
#pragma unroll
    for (int rt = 0; rt < 2; ++rt) {
      const _Float16* pa = As + bo + (wr * 32 + rt * 16 + lo) * KS + hi * 8;
      af[rt] = ld_frag2(pa, pa + 16);
    }
#pragma unroll
    for (int ct = 0; ct < 4; ++ct) {
      const _Float16* pb = Bs + bo + (wc * 64 + ct * 16 + lo) * KS + hi * 8;
      bf[ct] = ld_frag2(pb, pb + 16);
    }
#pragma unroll
    for (int rt = 0; rt < 2; ++rt)
#pragma unroll
      for (int ct = 0; ct < 4; ++ct)
        acc[rt][ct] = __builtin_amdgcn_wmma_f32_16x16x32_f16(
            false, af[rt], false, bf[ct], (short)0, acc[rt][ct], false, false);

    buf ^= 1;
  }

  // ---- epilogue (C/D layout: col = lo, row = hi*8 + r) ------------------
#pragma unroll
  for (int rt = 0; rt < 2; ++rt) {
    const int rbase = m0 + wr * 32 + rt * 16 + hi * 8;
#pragma unroll
    for (int ct = 0; ct < 4; ++ct) {
      const int col = n0 + wc * 64 + ct * 16 + lo;
      const float bb = bias ? bias[col] : 0.0f;
      v8f c = acc[rt][ct];
      if (mode == MODE_MEAN4) {
        const float s0 = c[0] + c[1] + c[2] + c[3];
        const float s1 = c[4] + c[5] + c[6] + c[7];
        const int g0 = rbase >> 2;            // rbase is a multiple of 8
        Y[(size_t)g0 * N + col]       = s0 * 0.25f + bb;
        Y[(size_t)(g0 + 1) * N + col] = s1 * 0.25f + bb;
      } else if (mode == MODE_VEC) {
        const float ev = extra[col];
#pragma unroll
        for (int r = 0; r < 8; ++r)
          Y[(size_t)(rbase + r) * N + col] = c[r] + bb + ev;
      } else if (mode == MODE_RELU) {
#pragma unroll
        for (int r = 0; r < 8; ++r)
          Y[(size_t)(rbase + r) * N + col] = fmaxf(c[r] + bb, 0.0f);
      } else if (mode == MODE_SIG) {
#pragma unroll
        for (int r = 0; r < 8; ++r) {
          const float v = c[r] + bb;
          Y[(size_t)(rbase + r) * N + col] = 1.0f / (1.0f + __expf(-v));
        }
      } else {
#pragma unroll
        for (int r = 0; r < 8; ++r)
          Y[(size_t)(rbase + r) * N + col] = c[r] + bb;
      }
    }
  }
}

// ---------------- tiny attention core (seq = E = 4, d = 128) ------------
// One thread per (b, head, query-row). KV rows: [k(1024) | v(1024)].
__global__ __launch_bounds__(256)
void misa_attn(const float* __restrict__ Q, const float* __restrict__ KV,
               float* __restrict__ O)
{
  const int t = blockIdx.x * 256 + threadIdx.x;
  if (t >= BATCH * NH * EF) return;
  const int eq = t & (EF - 1);
  const int hh = (t >> 2) & (NH - 1);
  const int b  = t >> 5;
  const float* qp = Q + (size_t)(b * EF + eq) * HD + hh * DH;
  const float* kp[EF];
  const float* vp[EF];
#pragma unroll
  for (int e = 0; e < EF; ++e) {
    kp[e] = KV + (size_t)(b * EF + e) * (2 * HD) + hh * DH;
    vp[e] = kp[e] + HD;
  }
  float sc[EF] = {0.f, 0.f, 0.f, 0.f};
  for (int j = 0; j < DH; j += 4) {
    const float4 qv = *(const float4*)(qp + j);
#pragma unroll
    for (int e = 0; e < EF; ++e) {
      const float4 kv4 = *(const float4*)(kp[e] + j);
      sc[e] += qv.x * kv4.x + qv.y * kv4.y + qv.z * kv4.z + qv.w * kv4.w;
    }
  }
  const float scale = 0.08838834764831845f;   // 1/sqrt(128)
  float mx = -1e30f;
#pragma unroll
  for (int e = 0; e < EF; ++e) { sc[e] *= scale; mx = fmaxf(mx, sc[e]); }
  float se = 0.f;
#pragma unroll
  for (int e = 0; e < EF; ++e) { sc[e] = __expf(sc[e] - mx); se += sc[e]; }
  const float inv = 1.0f / se;
#pragma unroll
  for (int e = 0; e < EF; ++e) sc[e] *= inv;
  float* op = O + (size_t)(b * EF + eq) * HD + hh * DH;
  for (int j = 0; j < DH; j += 4) {
    float4 o = {0.f, 0.f, 0.f, 0.f};
#pragma unroll
    for (int e = 0; e < EF; ++e) {
      const float4 vv = *(const float4*)(vp[e] + j);
      o.x += sc[e] * vv.x; o.y += sc[e] * vv.y;
      o.z += sc[e] * vv.z; o.w += sc[e] * vv.w;
    }
    *(float4*)(op + j) = o;
  }
}

// ---------------- LayerNorm(X + R) -> Y (1024 cols, block = row) --------
__global__ __launch_bounds__(256)
void misa_ln_res(const float* __restrict__ X, const float* __restrict__ R,
                 const float* __restrict__ g, const float* __restrict__ b,
                 float* __restrict__ Y)
{
  __shared__ float s1[256], s2[256];
  const int row = blockIdx.x, tid = threadIdx.x;
  const float* x = X + (size_t)row * HD;
  const float* r = R + (size_t)row * HD;
  float v[4]; float a1 = 0.f, a2 = 0.f;
#pragma unroll
  for (int i = 0; i < 4; ++i) {
    const int c = tid + i * 256;
    v[i] = x[c] + r[c];
    a1 += v[i]; a2 += v[i] * v[i];
  }
  s1[tid] = a1; s2[tid] = a2; __syncthreads();
  for (int st = 128; st > 0; st >>= 1) {
    if (tid < st) { s1[tid] += s1[tid + st]; s2[tid] += s2[tid + st]; }
    __syncthreads();
  }
  const float mean = s1[0] * (1.0f / HD);
  const float var  = s2[0] * (1.0f / HD) - mean * mean;
  const float rstd = rsqrtf(var + EPSLN);
  float* y = Y + (size_t)row * HD;
#pragma unroll
  for (int i = 0; i < 4; ++i) {
    const int c = tid + i * 256;
    y[c] = (v[i] - mean) * rstd * g[c] + b[c];
  }
}

// ---------------- final: LN(X) then residual mix -> out ------------------
__global__ __launch_bounds__(256)
void misa_final(const float* __restrict__ X, const float* __restrict__ tf,
                const float* __restrict__ sf, const float* __restrict__ g,
                const float* __restrict__ b, const float* __restrict__ rw,
                float* __restrict__ Y)
{
  __shared__ float s1[256], s2[256];
  const int row = blockIdx.x, tid = threadIdx.x;
  const float* x = X + (size_t)row * HD;
  float v[4]; float a1 = 0.f, a2 = 0.f;
#pragma unroll
  for (int i = 0; i < 4; ++i) {
    const int c = tid + i * 256;
    v[i] = x[c];
    a1 += v[i]; a2 += v[i] * v[i];
  }
  s1[tid] = a1; s2[tid] = a2; __syncthreads();
  for (int st = 128; st > 0; st >>= 1) {
    if (tid < st) { s1[tid] += s1[tid + st]; s2[tid] += s2[tid + st]; }
    __syncthreads();
  }
  const float mean = s1[0] * (1.0f / HD);
  const float var  = s2[0] * (1.0f / HD) - mean * mean;
  const float rstd = rsqrtf(var + EPSLN);
  const float w = rw[0];
  float* y = Y + (size_t)row * HD;
#pragma unroll
  for (int i = 0; i < 4; ++i) {
    const int c = tid + i * 256;
    const float ln = (v[i] - mean) * rstd * g[c] + b[c];
    const float orig = 0.5f * (tf[(size_t)row * HD + c] + sf[(size_t)row * HD + c]);
    y[c] = w * ln + (1.0f - w) * orig;
  }
}

// ---------------- small elementwise helpers ------------------------------
__global__ void misa_meanpair(const float* __restrict__ T, const float* __restrict__ S,
                              float* __restrict__ O)   // O[B, 2H] = [mean_E T | mean_E S]
{
  const int t = blockIdx.x * 256 + threadIdx.x;
  if (t >= BATCH * 2 * HD) return;
  const int c = t & (2 * HD - 1);
  const int b = t >> 11;
  const float* src = (c < HD) ? T : S;
  const int cc = c & (HD - 1);
  float s = 0.f;
#pragma unroll
  for (int e = 0; e < EF; ++e) s += src[(size_t)(b * EF + e) * HD + cc];
  O[t] = s * 0.25f;
}

__global__ void misa_bcast(const float* __restrict__ J, float* __restrict__ O)
{ // O[(b*E+e)*HD + c] = J[b*HD + c]
  const int t = blockIdx.x * 256 + threadIdx.x;
  if (t >= BE * HD) return;
  const int c = t & (HD - 1);
  const int row = t >> 10;
  O[t] = J[(size_t)(row >> 2) * HD + c];
}

__global__ void misa_concat(const float* __restrict__ A, const float* __restrict__ Bp,
                            float* __restrict__ O)
{ // O[b, 0:H] = A[b], O[b, H:2H] = Bp[b]
  const int t = blockIdx.x * 256 + threadIdx.x;
  if (t >= BATCH * 2 * HD) return;
  const int c = t & (2 * HD - 1);
  const int b = t >> 11;
  O[t] = (c < HD) ? A[(size_t)b * HD + c] : Bp[(size_t)b * HD + (c - HD)];
}

__global__ void misa_fusedbuild(const float* __restrict__ GT, const float* __restrict__ GS,
                                const float* __restrict__ GJ,
                                const float* __restrict__ MTS, const float* __restrict__ MST,
                                const float* __restrict__ MTJ, const float* __restrict__ MSJ,
                                const float* __restrict__ MJT, const float* __restrict__ MJS,
                                float* __restrict__ F)
{
  const int t = blockIdx.x * 256 + threadIdx.x;
  if (t >= BATCH * HD) return;
  const int c = t & (HD - 1);
  const int b = t >> 10;
  const float gt = GT[t], gs = GS[t], gj = GJ[t];
  const size_t base = (size_t)b * (6 * HD) + c;
  F[base + 0 * HD] = gt * MTS[t];
  F[base + 1 * HD] = gs * MST[t];
  F[base + 2 * HD] = gt * MTJ[t];
  F[base + 3 * HD] = gs * MSJ[t];
  F[base + 4 * HD] = gj * MJT[t];
  F[base + 5 * HD] = gj * MJS[t];
}

// ---------------- orchestration ------------------------------------------
extern "C" void kernel_launch(void* const* d_in, const int* in_sizes, int n_in,
                              void* d_out, int out_size, void* d_ws, size_t ws_size,
                              hipStream_t stream)
{
  const float* temporal = (const float*)d_in[0];
  const float* spatial  = (const float*)d_in[1];
  const float* exp_t_w  = (const float*)d_in[2];
  const float* exp_t_b  = (const float*)d_in[3];
  const float* exp_s_w  = (const float*)d_in[4];
  const float* exp_s_b  = (const float*)d_in[5];
  const float* pos_enc  = (const float*)d_in[6];
  const float* mha_in_w = (const float*)d_in[7];
  const float* mha_in_b = (const float*)d_in[8];
  const float* mha_out_w= (const float*)d_in[9];
  const float* mha_out_b= (const float*)d_in[10];
  const float* joint_w  = (const float*)d_in[11];
  const float* joint_b  = (const float*)d_in[12];
  const float* gate_w   = (const float*)d_in[13];
  const float* gate_b   = (const float*)d_in[14];
  const float* out1_w   = (const float*)d_in[15];
  const float* out1_b   = (const float*)d_in[16];
  const float* out2_w   = (const float*)d_in[17];
  const float* out2_b   = (const float*)d_in[18];
  const float* ln_g     = (const float*)d_in[19];
  const float* ln_b     = (const float*)d_in[20];
  const float* res_w    = (const float*)d_in[21];
  float* out            = (float*)d_out;
  (void)in_sizes; (void)n_in; (void)out_size; (void)ws_size;

  // ---- workspace layout (f32), with aliasing once producers retire -----
  float* ws = (float*)d_ws;
  size_t off = 0;
  auto take = [&](size_t n) { float* p = ws + off; off += n; return p; };
  const size_t R16 = (size_t)BE * HD;            // 16384 x 1024
  float* TE  = take(R16);                        // t_exp -> t_enh (in place)
  float* SE  = take(R16);                        // s_exp -> s_enh (in place)
  float* Qb  = take(R16);                        // q projections
  float* KVb = take((size_t)BE * 2 * HD);        // [k|v] projections
  float* AOb = take(R16);                        // attention output (pre out-proj)
  float* COb = take(R16);                        // self-attn out-proj (pre-LN)
  float* JE  = take(R16);                        // j_exp broadcast
  float* Jb  = take((size_t)BATCH * HD);         // joint
  float* MTS = take((size_t)BATCH * HD);
  float* MST = take((size_t)BATCH * HD);
  float* MTJ = take((size_t)BATCH * HD);
  float* MSJ = take((size_t)BATCH * HD);
  float* MJT = take((size_t)BATCH * HD);
  float* MJS = take((size_t)BATCH * HD);
  float* M2  = take((size_t)BATCH * 2 * HD);     // means / concat staging
  // aliases (producers dead by the time these are written):
  float* FUSED = KVb;                            // [4096, 6144] <= [16384, 2048]
  float* H1    = COb;                            // [4096, 2048] <= [16384, 1024]
  float* H2    = Qb;                             // [4096, 1024]
  float* GT    = JE;
  float* GS    = JE + (size_t)BATCH * HD;
  float* GJ    = JE + (size_t)2 * BATCH * HD;

  auto gemm = [&](const float* A, const float* W, const float* bias, const float* extra,
                  float* Y, int M, int N, int K, int mode) {
    dim3 grid(N / 128, M / 128);
    misa_gemm_wmma<<<grid, 256, 0, stream>>>(A, W, bias, extra, Y, M, N, K, mode);
  };
  auto ew_blocks = [](size_t n) { return (unsigned)((n + 255) / 256); };
  auto mha_qkv_attn = [&](const float* q, const float* kv, int i) {
    const float* wq  = mha_in_w + (size_t)i * 3 * HD * HD;
    const float* wkv = wq + (size_t)HD * HD;
    const float* bq  = mha_in_b + (size_t)i * 3 * HD;
    const float* bkv = bq + HD;
    gemm(q,  wq,  bq,  nullptr, Qb,  BE, HD,     HD, MODE_BIAS);
    gemm(kv, wkv, bkv, nullptr, KVb, BE, 2 * HD, HD, MODE_BIAS);
    misa_attn<<<ew_blocks((size_t)BATCH * NH * EF), 256, 0, stream>>>(Qb, KVb, AOb);
  };
  auto mha_out = [&](int i, float* Y, int mode) {
    gemm(AOb, mha_out_w + (size_t)i * HD * HD, mha_out_b + (size_t)i * HD,
         nullptr, Y, BE, HD, HD, mode);
  };

  // ---- phase A: expansion (+ bias + pos_enc fused) ----------------------
  gemm(temporal, exp_t_w, exp_t_b, pos_enc, TE, BATCH, EF * HD, HD, MODE_VEC);
  gemm(spatial,  exp_s_w, exp_s_b, pos_enc, SE, BATCH, EF * HD, HD, MODE_VEC);

  // ---- phase B: self attentions + residual LN ---------------------------
  mha_qkv_attn(TE, TE, 0);
  mha_out(0, COb, MODE_BIAS);
  misa_ln_res<<<BE, 256, 0, stream>>>(COb, TE, ln_g + 0 * HD, ln_b + 0 * HD, TE);
  mha_qkv_attn(SE, SE, 1);
  mha_out(1, COb, MODE_BIAS);
  misa_ln_res<<<BE, 256, 0, stream>>>(COb, SE, ln_g + 1 * HD, ln_b + 1 * HD, SE);

  // ---- phase C: joint pathway -------------------------------------------
  misa_meanpair<<<ew_blocks((size_t)BATCH * 2 * HD), 256, 0, stream>>>(TE, SE, M2);
  gemm(M2, joint_w, joint_b, nullptr, Jb, BATCH, HD, 2 * HD, MODE_BIAS);
  misa_bcast<<<ew_blocks(R16), 256, 0, stream>>>(Jb, JE);

  // ---- phase D: six cross attentions (mean-over-E fused in out-proj) ----
  mha_qkv_attn(TE, SE, 2);  mha_out(2, MTS, MODE_MEAN4);
  mha_qkv_attn(SE, TE, 3);  mha_out(3, MST, MODE_MEAN4);
  mha_qkv_attn(TE, JE, 4);  mha_out(4, MTJ, MODE_MEAN4);
  mha_qkv_attn(SE, JE, 4);  mha_out(4, MSJ, MODE_MEAN4);
  mha_qkv_attn(JE, TE, 4);  mha_out(4, MJT, MODE_MEAN4);
  mha_qkv_attn(JE, SE, 4);  mha_out(4, MJS, MODE_MEAN4);

  // ---- phase E: gates (sigmoid fused), fusion MLP, final LN + mix -------
  misa_concat<<<ew_blocks((size_t)BATCH * 2 * HD), 256, 0, stream>>>(MTS, MTJ, M2);
  gemm(M2, gate_w + (size_t)0 * HD * 2 * HD, gate_b + 0 * HD, nullptr, GT, BATCH, HD, 2 * HD, MODE_SIG);
  misa_concat<<<ew_blocks((size_t)BATCH * 2 * HD), 256, 0, stream>>>(MST, MSJ, M2);
  gemm(M2, gate_w + (size_t)1 * HD * 2 * HD, gate_b + 1 * HD, nullptr, GS, BATCH, HD, 2 * HD, MODE_SIG);
  misa_concat<<<ew_blocks((size_t)BATCH * 2 * HD), 256, 0, stream>>>(MJT, MJS, M2);
  gemm(M2, gate_w + (size_t)2 * HD * 2 * HD, gate_b + 2 * HD, nullptr, GJ, BATCH, HD, 2 * HD, MODE_SIG);

  misa_fusedbuild<<<ew_blocks((size_t)BATCH * HD), 256, 0, stream>>>(
      GT, GS, GJ, MTS, MST, MTJ, MSJ, MJT, MJS, FUSED);
  gemm(FUSED, out1_w, out1_b, nullptr, H1, BATCH, 2 * HD, 6 * HD, MODE_RELU);
  gemm(H1,    out2_w, out2_b, nullptr, H2, BATCH, HD,     2 * HD, MODE_BIAS);
  misa_final<<<BATCH, 256, 0, stream>>>(H2, temporal, spatial,
                                        ln_g + 2 * HD, ln_b + 2 * HD, res_w, out);
}
